// MultiheadAttention_4612794876661
// MI455X (gfx1250) — matrix operand order repaired and checked by reading, not compiled, not run on hardware
//
#include <hip/hip_runtime.h>

// ---------------------------------------------------------------------------
// Causal multi-head self-attention (T=2048, B=2, D=1024, H=16, hd=64) for
// MI455X / gfx1250.  All GEMM work uses v_wmma_f32_16x16x32_bf16; projection
// GEMMs stage weight tiles through LDS with global_load_async_to_lds_b128
// (ASYNCcnt) so the 8 waves of a block share one copy of each B tile.
// Outputs: out [T,B,D] fp32, then avg attention weights [B,T,T] fp32.
// ---------------------------------------------------------------------------

typedef __bf16 bf16_t;
typedef __attribute__((ext_vector_type(8)))  __bf16 bf16x8;
typedef __attribute__((ext_vector_type(16))) __bf16 bf16x16;
typedef __attribute__((ext_vector_type(8)))  float  f32x8;

#define T_DIM 2048
#define B_DIM 2
#define D_DIM 1024
#define H_DIM 16
#define HD    64
#define NROWS (T_DIM * B_DIM) /* 4096 GEMM rows, row = t*B + b */

// ---- fragment helpers ------------------------------------------------------
// A-matrix (16x32 bf16) per-lane layout used here (contiguous K, not the ISA order): lanes 0-15 hold
// K = {0..15}, lanes 16-31 hold K = {16..31} (matches B)  -> two 16B loads.
__device__ __forceinline__ bf16x16 make_afrag(const bf16_t* p0, const bf16_t* p1) {
  bf16x8 lo = *(const bf16x8*)p0;
  bf16x8 hi = *(const bf16x8*)p1;
  bf16x16 r;
#pragma unroll
  for (int i = 0; i < 8; ++i) { r[i] = lo[i]; r[i + 8] = hi[i]; }
  return r;
}

// B-matrix (32x16 bf16): lane l holds column n = l%16, K chunk (l/16)*16..+15
// -> one contiguous 32B load (caller computes the address).

__device__ __forceinline__ f32x8 wmma_bf16(bf16x16 a, bf16x16 b, f32x8 c) {
  return __builtin_amdgcn_wmma_f32_16x16x32_bf16(
      /*neg_a=*/false, a, /*neg_b=*/false, b,
      /*c_mod=*/(short)0, c, /*reuse_a=*/false, /*reuse_b=*/false);
}

// ---- CDNA5 async-to-LDS helpers (inline asm; tracked on ASYNCcnt) ----------
__device__ __forceinline__ void async_to_lds_b128(unsigned lds_addr,
                                                  const void* gaddr) {
  asm volatile("global_load_async_to_lds_b128 %0, %1, off"
               :
               : "v"(lds_addr), "v"(gaddr)
               : "memory");
}
__device__ __forceinline__ void wait_async_le1() {
  asm volatile("s_wait_asynccnt 0x1" ::: "memory");
}
__device__ __forceinline__ void wait_async_0() {
  asm volatile("s_wait_asynccnt 0x0" ::: "memory");
}

// ---- kernel 0: fp32 -> bf16 conversion ------------------------------------
__global__ void cvt_f32_bf16_kernel(const float* __restrict__ src,
                                    unsigned short* __restrict__ dst, int n) {
  int i = blockIdx.x * 256 + threadIdx.x;
  if (i < n) ((bf16_t*)dst)[i] = (bf16_t)src[i];
}

// ---- kernel 1: fused QKV projection ---------------------------------------
// out = X @ W^T + bias (Q additionally * 1/sqrt(hd)).  grid.z = 0/1/2 (Q/K/V).
// Each wave: 16 (rows) x 64 (cols) tile; K loop over 1024 in steps of 32.
// The 64x32 bf16 weight tile (4KB) is async-staged into LDS, double buffered.
// Q,K stored [B,H,T,hd] bf16; V stored transposed [B,H,hd,T] bf16.
__global__ void __launch_bounds__(256)
qkv_proj_kernel(const unsigned short* __restrict__ Xu,
                const unsigned short* __restrict__ Wqu,
                const unsigned short* __restrict__ Wku,
                const unsigned short* __restrict__ Wvu,
                const float* __restrict__ qb, const float* __restrict__ kb,
                const float* __restrict__ vb,
                unsigned short* __restrict__ Qu, unsigned short* __restrict__ Ku,
                unsigned short* __restrict__ VTu) {
  __shared__ unsigned short btile[2][64 * 32]; // [buf][col][k] bf16, 2x4KB

  const bf16_t* X = (const bf16_t*)Xu;
  const int z = blockIdx.z;
  const bf16_t* W = (z == 0) ? (const bf16_t*)Wqu
                  : (z == 1) ? (const bf16_t*)Wku : (const bf16_t*)Wvu;
  const float* bias = (z == 0) ? qb : (z == 1) ? kb : vb;
  const float scale = (z == 0) ? 0.125f : 1.0f; // 1/sqrt(64)

  const int tid = threadIdx.x;
  const int w = tid >> 5, lane = tid & 31;
  const int lhi = lane >> 4, llo = lane & 15;
  const int m0 = blockIdx.x * 128 + w * 16; // row tile
  const int n0 = blockIdx.y * 64;           // col tile

  // staging map: thread -> one 16B chunk of the 64x32 tile
  const int scol = tid >> 2;          // 0..63
  const int skp  = (tid & 3) * 8;     // 0,8,16,24

  // prologue: stage tile for k0 = 0 into buffer 0
  async_to_lds_b128((unsigned)(unsigned long long)&btile[0][scol * 32 + skp],
                    W + (size_t)(n0 + scol) * D_DIM + skp);

  f32x8 acc[4] = {};
  const bf16_t* arow = X + (size_t)(m0 + llo) * D_DIM;
  for (int k0 = 0; k0 < D_DIM; k0 += 32) {
    const int buf = (k0 >> 5) & 1;
    const bool more = (k0 + 32) < D_DIM;
    if (more)
      async_to_lds_b128(
          (unsigned)(unsigned long long)&btile[buf ^ 1][scol * 32 + skp],
          W + (size_t)(n0 + scol) * D_DIM + (k0 + 32) + skp);
    if (more) wait_async_le1(); else wait_async_0();
    __syncthreads();

    const bf16_t* ap = arow + k0 + lhi * 16;
    bf16x16 a = make_afrag(ap, ap + 8);
#pragma unroll
    for (int j = 0; j < 4; ++j) {
      const bf16_t* bp = (const bf16_t*)&btile[buf][(j * 16 + llo) * 32 + lhi * 16];
      acc[j] = wmma_bf16(a, *(const bf16x16*)bp, acc[j]);
    }
    __syncthreads(); // protect buf from being re-staged while in use
  }

#pragma unroll
  for (int j = 0; j < 4; ++j) {
#pragma unroll
    for (int r = 0; r < 8; ++r) {
      int row = m0 + r + lhi * 8;       // = t*B + b
      int col = n0 + j * 16 + llo;      // = h*hd + d
      float v = (acc[j][r] + bias[col]) * scale;
      int t = row >> 1, b = row & 1;
      int h = col >> 6, d = col & 63;
      if (z == 2) {
        ((bf16_t*)VTu)[((size_t)(b * H_DIM + h) * HD + d) * T_DIM + t] = (bf16_t)v;
      } else {
        bf16_t* dst = (z == 0) ? (bf16_t*)Qu : (bf16_t*)Ku;
        dst[((size_t)(b * H_DIM + h) * T_DIM + t) * HD + d] = (bf16_t)v;
      }
    }
  }
}

// ---- kernel 2: causal attention -------------------------------------------
// One workgroup (8 waves) = one (batch, 16-row t tile); loops over all heads
// so it exclusively owns avg_weights[b, t0:t0+16, :] (no atomics needed).
// Dynamic LDS: bf16 scores/probs 16x2048 (64KB) + fp32 avg 16x2048 (128KB)
//            + fp32 partial-O 8x16x64 (32KB) = 224KB of the WGP's 320KB.
#define SPITCH 2048
__global__ void __launch_bounds__(256)
attn_kernel(const unsigned short* __restrict__ Qu,
            const unsigned short* __restrict__ Ku,
            const unsigned short* __restrict__ VTu,
            unsigned short* __restrict__ Attnu,
            float* __restrict__ avgG) {
  extern __shared__ char smem[];
  bf16_t* sc   = (bf16_t*)smem;                                  // 16*2048 bf16
  float*  avgL = (float*)(smem + 16 * SPITCH * 2);               // 16*2048 f32
  float*  red  = (float*)(smem + 16 * SPITCH * 2 + 16 * SPITCH * 4); // 8*1024

  const int t0 = blockIdx.x * 16;
  const int b  = blockIdx.y;
  const int tid = threadIdx.x;
  const int w = tid >> 5, lane = tid & 31;
  const int lhi = lane >> 4, llo = lane & 15;

  const int nS = t0 / 16 + 1;               // causal s-blocks of 16
  const int sLen = nS * 16;
  const int sLen32 = (sLen + 31) & ~31;     // pad to WMMA K granularity

  for (int i = tid; i < 16 * SPITCH; i += 256) avgL[i] = 0.0f;

  for (int h = 0; h < H_DIM; ++h) {
    const bf16_t* Qh = (const bf16_t*)Qu + (size_t)(b * H_DIM + h) * T_DIM * HD;
    const bf16_t* Kh = (const bf16_t*)Ku + (size_t)(b * H_DIM + h) * T_DIM * HD;
    const bf16_t* Vh = (const bf16_t*)VTu + (size_t)(b * H_DIM + h) * HD * T_DIM;

    // zero the 16-column pad (probs there must be 0 for the P@V WMMAs)
    if (sLen32 != sLen) {
      int m = tid >> 4, n = tid & 15; // 256 threads cover 16x16 exactly
      sc[m * SPITCH + sLen + n] = (bf16_t)0.0f;
    }

    // ---- scores S = Q K^T, causal mask, -> bf16 LDS ----
    const bf16_t* qrow = Qh + (size_t)(t0 + llo) * HD;
    bf16x16 qa0 = make_afrag(qrow + lhi * 16, qrow + lhi * 16 + 8);
    bf16x16 qa1 = make_afrag(qrow + 32 + lhi * 16, qrow + 32 + lhi * 16 + 8);
    for (int j = w; j < nS; j += 8) {
      int s0 = j * 16;
      f32x8 c = {};
      const bf16_t* kp0 = Kh + (size_t)(s0 + llo) * HD + lhi * 16;
      c = wmma_bf16(qa0, *(const bf16x16*)kp0, c);
      c = wmma_bf16(qa1, *(const bf16x16*)(kp0 + 32), c);
#pragma unroll
      for (int r = 0; r < 8; ++r) {
        int m = r + lhi * 8;
        float v = c[r];
        if (s0 + llo > t0 + m) v = -1e30f; // causal mask (diag block only)
        sc[m * SPITCH + s0 + llo] = (bf16_t)v;
      }
    }
    __syncthreads();

    // ---- exact row softmax; probs overwrite scores; accumulate avg ----
    {
      int m = 2 * w + lhi;             // wave w owns rows 2w, 2w+1
      int base = m * SPITCH;
      float mx = -1e30f;
      for (int s = llo; s < sLen; s += 16) mx = fmaxf(mx, (float)sc[base + s]);
#pragma unroll
      for (int o = 1; o < 16; o <<= 1) mx = fmaxf(mx, __shfl_xor(mx, o, 16));
      float sum = 0.0f;
      for (int s = llo; s < sLen; s += 16) sum += __expf((float)sc[base + s] - mx);
#pragma unroll
      for (int o = 1; o < 16; o <<= 1) sum += __shfl_xor(sum, o, 16);
      float inv = 1.0f / sum;
      for (int s = llo; s < sLen; s += 16) {
        float p = __expf((float)sc[base + s] - mx) * inv;
        sc[base + s] = (bf16_t)p;
        avgL[base + s] += p * (1.0f / (float)H_DIM);
      }
    }
    __syncthreads();

    // ---- O = P @ V : each wave takes strided K(=s) chunks of 32 ----
    f32x8 oacc[4] = {};
    for (int kb = w * 32; kb < sLen32; kb += 256) {
      const bf16_t* pr = sc + (size_t)llo * SPITCH + kb + lhi * 16;
      bf16x16 pa = make_afrag(pr, pr + 8);
#pragma unroll
      for (int dblk = 0; dblk < 4; ++dblk) {
        const bf16_t* vp = Vh + (size_t)(dblk * 16 + llo) * T_DIM + kb + lhi * 16;
        oacc[dblk] = wmma_bf16(pa, *(const bf16x16*)vp, oacc[dblk]);
      }
    }
#pragma unroll
    for (int dblk = 0; dblk < 4; ++dblk)
#pragma unroll
      for (int r = 0; r < 8; ++r) {
        int m = r + lhi * 8;
        red[w * 1024 + m * 64 + dblk * 16 + llo] = oacc[dblk][r];
      }
    __syncthreads();

    // ---- reduce the 8 wave partials, store attn head tile (bf16) ----
    for (int i = tid; i < 1024; i += 256) {
      float s = 0.0f;
#pragma unroll
      for (int ww = 0; ww < 8; ++ww) s += red[ww * 1024 + i];
      int m = i >> 6, d = i & 63;
      ((bf16_t*)Attnu)[((size_t)(b * H_DIM + h) * T_DIM + (t0 + m)) * HD + d] =
          (bf16_t)s;
    }
    __syncthreads();
  }

  // ---- write avg attention weights [B,T,T] (zeros past causal bound) ----
  for (int i = tid; i < 16 * SPITCH; i += 256) {
    int m = i >> 11, s = i & 2047;
    avgG[((size_t)(b * T_DIM + t0 + m)) * T_DIM + s] = avgL[i];
  }
}

// ---- kernel 3: output projection ------------------------------------------
// out[t,b,:] = attn_row @ Wo^T + out_b  (fp32 output straight to d_out)
// Same async-staged LDS double buffer for the Wo tile as the QKV kernel.
__global__ void __launch_bounds__(256)
out_proj_kernel(const unsigned short* __restrict__ Attnu,
                const unsigned short* __restrict__ Wou,
                const float* __restrict__ ob, float* __restrict__ out) {
  __shared__ unsigned short btile[2][64 * 32];

  const bf16_t* Attn = (const bf16_t*)Attnu;
  const bf16_t* Wo = (const bf16_t*)Wou;
  const int tid = threadIdx.x;
  const int w = tid >> 5, lane = tid & 31;
  const int lhi = lane >> 4, llo = lane & 15;
  const int m0 = blockIdx.x * 128 + w * 16;
  const int n0 = blockIdx.y * 64;

  const int scol = tid >> 2;
  const int skp  = (tid & 3) * 8;

  async_to_lds_b128((unsigned)(unsigned long long)&btile[0][scol * 32 + skp],
                    Wo + (size_t)(n0 + scol) * D_DIM + skp);

  int row = m0 + llo;
  int t = row >> 1, b = row & 1;

  f32x8 acc[4] = {};
  for (int k0 = 0; k0 < D_DIM; k0 += 32) {
    const int buf = (k0 >> 5) & 1;
    const bool more = (k0 + 32) < D_DIM;
    if (more)
      async_to_lds_b128(
          (unsigned)(unsigned long long)&btile[buf ^ 1][scol * 32 + skp],
          Wo + (size_t)(n0 + scol) * D_DIM + (k0 + 32) + skp);
    if (more) wait_async_le1(); else wait_async_0();
    __syncthreads();

    int off1 = k0 + lhi * 16;  // 8-element chunks never straddle a head (hd=64)
    int off2 = off1 + 8;
    const bf16_t* p1 =
        Attn + ((size_t)(b * H_DIM + (off1 >> 6)) * T_DIM + t) * HD + (off1 & 63);
    const bf16_t* p2 =
        Attn + ((size_t)(b * H_DIM + (off2 >> 6)) * T_DIM + t) * HD + (off2 & 63);
    bf16x16 a = make_afrag(p1, p2);
#pragma unroll
    for (int j = 0; j < 4; ++j) {
      const bf16_t* bp = (const bf16_t*)&btile[buf][(j * 16 + llo) * 32 + lhi * 16];
      acc[j] = wmma_bf16(a, *(const bf16x16*)bp, acc[j]);
    }
    __syncthreads();
  }
#pragma unroll
  for (int j = 0; j < 4; ++j)
#pragma unroll
    for (int r = 0; r < 8; ++r) {
      int rr = m0 + r + lhi * 8;
      int col = n0 + j * 16 + llo;
      out[(size_t)rr * D_DIM + col] = acc[j][r] + ob[col];
    }
}

// ---------------------------------------------------------------------------
extern "C" void kernel_launch(void* const* d_in, const int* in_sizes, int n_in,
                              void* d_out, int out_size, void* d_ws, size_t ws_size,
                              hipStream_t stream) {
  (void)in_sizes; (void)n_in; (void)out_size; (void)ws_size;

  const float* query = (const float*)d_in[0];
  const float* q_w   = (const float*)d_in[1];
  const float* q_b   = (const float*)d_in[2];
  const float* k_w   = (const float*)d_in[3];
  const float* k_b   = (const float*)d_in[4];
  const float* v_w   = (const float*)d_in[5];
  const float* v_b   = (const float*)d_in[6];
  const float* out_w = (const float*)d_in[7];
  const float* out_b = (const float*)d_in[8];

  float* out = (float*)d_out;                               // [T,B,D]
  float* avg = out + (size_t)T_DIM * B_DIM * D_DIM;          // [B,T,T]

  // workspace layout (bf16 buffers), ~50 MB total
  char* ws = (char*)d_ws;
  size_t o = 0;
  unsigned short* Xb  = (unsigned short*)(ws + o); o += (size_t)NROWS * D_DIM * 2;
  unsigned short* Wqb = (unsigned short*)(ws + o); o += (size_t)D_DIM * D_DIM * 2;
  unsigned short* Wkb = (unsigned short*)(ws + o); o += (size_t)D_DIM * D_DIM * 2;
  unsigned short* Wvb = (unsigned short*)(ws + o); o += (size_t)D_DIM * D_DIM * 2;
  unsigned short* Wob = (unsigned short*)(ws + o); o += (size_t)D_DIM * D_DIM * 2;
  unsigned short* Qb  = (unsigned short*)(ws + o); o += (size_t)B_DIM * H_DIM * T_DIM * HD * 2;
  unsigned short* Kb  = (unsigned short*)(ws + o); o += (size_t)B_DIM * H_DIM * T_DIM * HD * 2;
  unsigned short* VTb = (unsigned short*)(ws + o); o += (size_t)B_DIM * H_DIM * HD * T_DIM * 2;
  unsigned short* Ab  = (unsigned short*)(ws + o); o += (size_t)B_DIM * H_DIM * T_DIM * HD * 2;

  const int nX = NROWS * D_DIM;       // 4,194,304
  const int nW = D_DIM * D_DIM;       // 1,048,576
  cvt_f32_bf16_kernel<<<(nX + 255) / 256, 256, 0, stream>>>(query, Xb, nX);
  cvt_f32_bf16_kernel<<<(nW + 255) / 256, 256, 0, stream>>>(q_w, Wqb, nW);
  cvt_f32_bf16_kernel<<<(nW + 255) / 256, 256, 0, stream>>>(k_w, Wkb, nW);
  cvt_f32_bf16_kernel<<<(nW + 255) / 256, 256, 0, stream>>>(v_w, Wvb, nW);
  cvt_f32_bf16_kernel<<<(nW + 255) / 256, 256, 0, stream>>>(out_w, Wob, nW);

  // QKV projections: grid (rows/128, cols/64, 3 matrices)
  qkv_proj_kernel<<<dim3(NROWS / 128, D_DIM / 64, 3), 256, 0, stream>>>(
      Xb, Wqb, Wkb, Wvb, q_b, k_b, v_b, Qb, Kb, VTb);

  // Attention: grid (T/16 row tiles, B), 224 KB dynamic LDS per workgroup
  const size_t lds_bytes =
      (size_t)16 * SPITCH * 2 + (size_t)16 * SPITCH * 4 + (size_t)8 * 1024 * 4;
  attn_kernel<<<dim3(T_DIM / 16, B_DIM), 256, lds_bytes, stream>>>(
      Qb, Kb, VTb, Ab, avg);

  // Output projection
  out_proj_kernel<<<dim3(NROWS / 128, D_DIM / 64), 256, 0, stream>>>(
      Ab, Wob, out_b, out);
}
